// HybridCNNGRUModel_27161373180667
// MI455X (gfx1250) — matrix operand order repaired and checked by reading, not compiled, not run on hardware
//
#include <hip/hip_runtime.h>
#include <cstdint>
#include <cstddef>

// ---------------------------------------------------------------------------
// HybridCNNGRUModel for MI455X (gfx1250).
// All matmuls: v_wmma_f32_16x16x32_f16 (fp16 in, fp32 accumulate),
// branch-free b128 fragment loads, compile-time K, 2-deep load pipeline.
// Decoder: persistent 32-WG kernel, grid-synced phases, L2-resident state.
// ---------------------------------------------------------------------------

typedef _Float16 v16h __attribute__((ext_vector_type(16)));
typedef _Float16 v8h  __attribute__((ext_vector_type(8)));
typedef float    v8f  __attribute__((ext_vector_type(8)));

constexpr int B_ = 64, S_ = 128, H_ = 256, T_ = 2048;
constexpr int NWG = 32;   // decoder workgroups

#define DEV __device__ __forceinline__

// ------------------------------- input index map ---------------------------
enum {
  IN_PPG = 0, IN_ACC, IN_EDA, IN_PREV,
  PPG_W1 = 4,  ACC_W1 = 12, EDA_W1 = 20, PREV_W1 = 28,
  FP_W = 36, FP_B, FP_G, FP_BE,
  L0F_WIH = 40, L0F_WHH, L0F_BIH, L0F_BHH,
  L0B_WIH = 44, L0B_WHH, L0B_BIH, L0B_BHH,
  L1F_WIH = 48, L1F_WHH, L1F_BIH, L1F_BHH,
  L1B_WIH = 52, L1B_WHH, L1B_BIH, L1B_BHH,
  EP_W = 56, EP_B, ATTN_W, ATTN_B, V_W,
  D0_WIH = 61, D0_WHH, D0_BIH, D0_BHH,
  D1_WIH = 65, D1_WHH, D1_BIH, D1_BHH,
  OP1_W = 69, OP1_B, OP_G, OP_BE, OP2_W, OP2_B, OP3_W, OP3_B
};

// ------------------------------- WMMA tile helpers -------------------------
// One 16x16 tile of C += A[16 x K] * W^T.  A row-major [M][lda] f16,
// W row-major [N][ldw] f16 with column offset wof (B[k][n] = W[n][wof+k]).
// A fragment/lane: two contiguous 8-half blocks (16B loads);
// B fragment/lane: one contiguous 16-half block (32B load).
// 2-deep software pipeline: chunk k+1 loads issue before chunk-k WMMA.
template <int K>
DEV v8f tile_mm(const _Float16* A, int lda, const _Float16* W, int ldw, int wof,
                int m0, int n0, v8f acc) {
  const int lane = threadIdx.x & 31;
  const int half = lane >> 4;
  const int q = lane & 15;
  const _Float16* arow = A + (size_t)(m0 + q) * lda + half * 16;
  const _Float16* wrow = W + (size_t)(n0 + q) * ldw + wof + half * 16;
  v8h alo = *(const v8h*)(arow);
  v8h ahi = *(const v8h*)(arow + 8);
  v16h bv = *(const v16h*)(wrow);
#pragma unroll
  for (int k0 = 0; k0 < K; k0 += 32) {
    v16h av = __builtin_shufflevector(alo, ahi, 0, 1, 2, 3, 4, 5, 6, 7,
                                      8, 9, 10, 11, 12, 13, 14, 15);
    v16h bcur = bv;
    if (k0 + 32 < K) {   // compile-time under full unroll
      alo = *(const v8h*)(arow + k0 + 32);
      ahi = *(const v8h*)(arow + k0 + 40);
      bv  = *(const v16h*)(wrow + k0 + 32);
    }
    acc = __builtin_amdgcn_wmma_f32_16x16x32_f16(false, av, false, bcur,
                                                 (short)0, acc, false, false);
  }
  return acc;
}

// Compile-time-specialized stores for the persistent kernels.
template <bool BIAS>
DEV void st_f32(float* C, int ldc, const float* bias, int m0, int n0, v8f acc) {
  const int lane = threadIdx.x & 31;
  const int half = lane >> 4;
  const int q = lane & 15;
  float bb = BIAS ? bias[n0 + q] : 0.f;
#pragma unroll
  for (int i = 0; i < 8; ++i) {
    float v = acc[i];
    if (BIAS) v += bb;
    C[(size_t)(m0 + half * 8 + i) * ldc + n0 + q] = v;
  }
}

DEV void st_f16_leaky(_Float16* Ch, int ldch, const float* bias, int m0, int n0,
                      v8f acc) {
  const int lane = threadIdx.x & 31;
  const int half = lane >> 4;
  const int q = lane & 15;
  float bb = bias[n0 + q];
#pragma unroll
  for (int i = 0; i < 8; ++i) {
    float v = acc[i] + bb;
    v = (v >= 0.f) ? v : 0.2f * v;
    Ch[(size_t)(m0 + half * 8 + i) * ldch + n0 + q] = (_Float16)v;
  }
}

// Generic runtime store (one-shot GEMM path only).
DEV void tile_st_any(float* C, int ldc, _Float16* Ch, int ldch, const float* bias,
                     int m0, int n0, v8f acc, int act) {
  const int lane = threadIdx.x & 31;
  const int half = lane >> 4;
  const int q = lane & 15;
  float bb = bias ? bias[n0 + q] : 0.f;
#pragma unroll
  for (int i = 0; i < 8; ++i) {
    float v = acc[i] + bb;
    if (act == 1) v = tanhf(v);
    else if (act == 2) v = (v >= 0.f) ? v : 0.2f * v;
    int m = m0 + half * 8 + i;
    if (C)  C[(size_t)m * ldc + n0 + q] = v;
    if (Ch) Ch[(size_t)m * ldch + n0 + q] = (_Float16)v;
  }
}

DEV float sigm(float x) { return 1.f / (1.f + __expf(-x)); }

// ------------------------------- generic GEMM kernel -----------------------
template <int K>
__global__ void __launch_bounds__(256)
k_gemm(const _Float16* A, int lda, const _Float16* W, int ldw, int wof,
       const float* bias, float* C, _Float16* Ch, int ldc,
       int M, int N, int act) {
  int wid = threadIdx.x >> 5;
  int tile = blockIdx.x * 8 + wid;
  int ntiles_n = N >> 4;
  int total = (M >> 4) * ntiles_n;
  if (tile >= total) return;
  int m0 = (tile / ntiles_n) << 4;
  int n0 = (tile % ntiles_n) << 4;
  v8f acc{};
  acc = tile_mm<K>(A, lda, W, ldw, wof, m0, n0, acc);
  tile_st_any(C, ldc, Ch, ldc, bias, m0, n0, acc, act);
}

// ------------------------------- small utility kernels ---------------------
__global__ void k_f2h(_Float16* dst, const float* src, int n) {
  int i = blockIdx.x * 256 + threadIdx.x;
  if (i < n) dst[i] = (_Float16)src[i];
}

__global__ void k_padw(_Float16* dst, const float* src, int N, int Ksrc, int Kdst) {
  int i = blockIdx.x * 256 + threadIdx.x;
  if (i >= N * Kdst) return;
  int k = i % Kdst, n = i / Kdst;
  dst[i] = (k < Ksrc) ? (_Float16)src[(size_t)n * Ksrc + k] : (_Float16)0.f;
}

// y = leaky((conv(x,w)+b)*g+be)  NCW / OIW
__global__ void k_conv(const float* in, const float* w, const float* bias,
                       const float* g, const float* be, float* out,
                       int B, int Cin, int Cout, int L, int KW, int pad) {
  int id = blockIdx.x * 256 + threadIdx.x;
  if (id >= B * Cout * L) return;
  int l = id % L, co = (id / L) % Cout, b = id / (L * Cout);
  float acc = bias[co];
  for (int ci = 0; ci < Cin; ++ci) {
    const float* xp = in + ((size_t)b * Cin + ci) * L;
    const float* wp = w + ((size_t)co * Cin + ci) * KW;
    for (int k = 0; k < KW; ++k) {
      int p = l + k - pad;
      if (p >= 0 && p < L) acc += xp[p] * wp[k];
    }
  }
  float v = acc * g[co] + be[co];
  out[id] = (v >= 0.f) ? v : 0.2f * v;
}

// conv + scale/shift + leaky + maxpool2 fused
__global__ void k_conv_mp(const float* in, const float* w, const float* bias,
                          const float* g, const float* be, float* out,
                          int B, int Cin, int Cout, int L, int KW, int pad) {
  int Lo = L >> 1;
  int id = blockIdx.x * 256 + threadIdx.x;
  if (id >= B * Cout * Lo) return;
  int lo = id % Lo, co = (id / Lo) % Cout, b = id / (Lo * Cout);
  float mx = -3.0e38f;
  for (int h = 0; h < 2; ++h) {
    int l = 2 * lo + h;
    float acc = bias[co];
    for (int ci = 0; ci < Cin; ++ci) {
      const float* xp = in + ((size_t)b * Cin + ci) * L;
      const float* wp = w + ((size_t)co * Cin + ci) * KW;
      for (int k = 0; k < KW; ++k) {
        int p = l + k - pad;
        if (p >= 0 && p < L) acc += xp[p] * wp[k];
      }
    }
    float v = acc * g[co] + be[co];
    v = (v >= 0.f) ? v : 0.2f * v;
    mx = fmaxf(mx, v);
  }
  out[id] = mx;
}

__global__ void k_avgpool(const float* in, float* out, int R, int Lin, int k) {
  int Lo = Lin / k;
  int id = blockIdx.x * 256 + threadIdx.x;
  if (id >= R * Lo) return;
  int lo = id % Lo, r = id / Lo;
  float s = 0.f;
  for (int i = 0; i < k; ++i) s += in[(size_t)r * Lin + lo * k + i];
  out[id] = s / (float)k;
}

__global__ void k_interp(const float* in, float* out, int R, int Lin, int Lout) {
  int id = blockIdx.x * 256 + threadIdx.x;
  if (id >= R * Lout) return;
  int o = id % Lout, r = id / Lout;
  float pos = (float)o * ((float)(Lin - 1) / (float)(Lout - 1));
  int lo = (int)floorf(pos);
  int hi = (lo + 1 < Lin) ? lo + 1 : Lin - 1;
  float w = pos - (float)lo;
  out[id] = in[(size_t)r * Lin + lo] * (1.f - w) + in[(size_t)r * Lin + hi] * w;
}

// feat[b][s][c] (f16) from branch-final [B][C][128] buffers
__global__ void k_assemble(const float* ppg, const float* acc, const float* eda,
                           const float* prev, _Float16* feat) {
  int id = blockIdx.x * 256 + threadIdx.x;
  if (id >= B_ * S_ * 288) return;
  int c = id % 288, s = (id / 288) % S_, b = id / (288 * S_);
  float v;
  if (c < 64)       v = ppg [((size_t)b * 64  + c)         * S_ + s];
  else if (c < 128) v = acc [((size_t)b * 64  + (c - 64))  * S_ + s];
  else if (c < 160) v = eda [((size_t)b * 32  + (c - 128)) * S_ + s];
  else              v = prev[((size_t)b * 128 + (c - 160)) * S_ + s];
  feat[id] = (_Float16)v;
}

// row LayerNorm -> f16
__global__ void k_ln(const float* y, const float* g, const float* be,
                     _Float16* outh, int R, int D) {
  int r = blockIdx.x * 256 + threadIdx.x;
  if (r >= R) return;
  const float* yr = y + (size_t)r * D;
  float mu = 0.f;
  for (int j = 0; j < D; ++j) mu += yr[j];
  mu /= (float)D;
  float vr = 0.f;
  for (int j = 0; j < D; ++j) { float d = yr[j] - mu; vr += d * d; }
  vr /= (float)D;
  float inv = rsqrtf(vr + 1e-5f);
  for (int j = 0; j < D; ++j)
    outh[(size_t)r * D + j] = (_Float16)((yr[j] - mu) * inv * g[j] + be[j]);
}

// ------------------------------- encoder scan ------------------------------
// One block per direction (blockIdx.x: 0=fwd, 1=bwd), 1024 threads (32 waves).
// gi (fp32, incl. bih) precomputed; per step: gh = h @ whh^T via WMMA with
// h staged f16 in LDS; gh scratch in global (L2).
__global__ void __launch_bounds__(1024)
k_enc_scan(const float* gi_f, const float* gi_b,
           const _Float16* whh_fh, const _Float16* whh_bh,
           const float* bhh_f, const float* bhh_b,
           float* gh_f, float* gh_b,
           _Float16* ysh, int ysld, _Float16* hT_out) {
  const int dir = blockIdx.x;
  const float*    gi  = dir ? gi_b : gi_f;
  const _Float16* whh = dir ? whh_bh : whh_fh;
  const float*    bhh = dir ? bhh_b : bhh_f;
  float*          gh  = dir ? gh_b : gh_f;
  const int ysof = dir ? H_ : 0;

  __shared__ __align__(32) _Float16 hsh[B_ * H_];   // 32 KB f16 copy of h
  const int tid = threadIdx.x;
  float hreg[16];
#pragma unroll
  for (int k2 = 0; k2 < 16; ++k2) {
    hreg[k2] = 0.f;
    hsh[tid + k2 * 1024] = (_Float16)0.f;
  }
  __syncthreads();

  const int wave = tid >> 5;
  for (int i = 0; i < S_; ++i) {
    int s = dir ? (S_ - 1 - i) : i;
    // gh = h @ whh^T   (M=64, N=768, K=256): 192 tiles over 32 waves
    for (int tile = wave; tile < 192; tile += 32) {
      int nt = tile % 48, mt = tile / 48;
      v8f acc{};
      acc = tile_mm<256>(hsh, H_, whh, H_, 0, mt * 16, nt * 16, acc);
      st_f32<false>(gh, 768, nullptr, mt * 16, nt * 16, acc);
    }
    __threadfence_block();
    __syncthreads();
    // GRU combine; each thread owns 16 (b,j) elements
#pragma unroll
    for (int k2 = 0; k2 < 16; ++k2) {
      int idx = tid + k2 * 1024;
      int b = idx >> 8, j = idx & 255;
      size_t gbase = ((size_t)b * S_ + s) * 768;
      float gir = gi[gbase + j];
      float giz = gi[gbase + 256 + j];
      float gin = gi[gbase + 512 + j];
      float ghr = gh[b * 768 + j]       + bhh[j];
      float ghz = gh[b * 768 + 256 + j] + bhh[256 + j];
      float ghn = gh[b * 768 + 512 + j] + bhh[512 + j];
      float r = sigm(gir + ghr);
      float z = sigm(giz + ghz);
      float n = tanhf(gin + r * ghn);
      float hn = (1.f - z) * n + z * hreg[k2];
      hreg[k2] = hn;
      hsh[idx] = (_Float16)hn;
      ysh[((size_t)b * S_ + s) * ysld + ysof + j] = (_Float16)hn;
      if (hT_out && i == S_ - 1) hT_out[b * 512 + ysof + j] = (_Float16)hn;
    }
    __syncthreads();
  }
}

// ------------------------------- decoder -----------------------------------
struct DecArgs {
  // f16 weights
  const _Float16 *attn_h, *whh0h, *whh1h, *wih0ph, *wih1h, *op1h, *op2h;
  // fp32 params
  const float *attn_b, *bih0, *bhh0, *bih1, *bhh1;
  const float *op1_b, *op_g, *op_be, *op2_b, *op3_w, *op3_b, *v_w;
  // encoder products
  const float *enc_out, *enc_proj;
  // state / scratch
  float *q, *gh0, *gh1, *gi0, *gi1, *y1, *h0, *h1;
  _Float16 *a0h;   // [64][288]: col0 = y_prev, 1..257 = ctx, pad zeros
  _Float16 *gc;    // [64][512]: 0..256 = h1, 256..512 = ctx
  _Float16 *y1h, *y2h, *h0h, *h1h;
  float *out;
  int* bar;
};

DEV void gsync(int* bar, int& n) {
  __threadfence();
  __syncthreads();
  n++;
  if (threadIdx.x == 0) {
    atomicAdd(bar, 1);
    while (__hip_atomic_load(bar, __ATOMIC_ACQUIRE, __HIP_MEMORY_SCOPE_AGENT) < n * NWG)
      __builtin_amdgcn_s_sleep(1);
    __threadfence();
  }
  __syncthreads();
}

__global__ void __launch_bounds__(256)
k_decoder(DecArgs A) {
  const int tid = threadIdx.x;
  const int gwave = blockIdx.x * 8 + (tid >> 5);
  int syncno = 0;
  __shared__ float sc[2][S_];
  int cid = __builtin_amdgcn_cluster_id_x();   // gfx1250 cluster state probe
  (void)cid;

  for (int t = 0; t < T_; ++t) {
    // ---- Phase A: q = h1@Wh^T+b (64 tiles), gh0 = h0@Whh0^T (192),
    //               gh1 = h1@Whh1^T (192): 448 wave-jobs
    for (int job = gwave; job < 448; job += 256) {
      v8f acc{};
      if (job < 64) {
        int nt = job & 15, mt = job >> 4;
        acc = tile_mm<256>(A.h1h, H_, A.attn_h, 512, 0, mt * 16, nt * 16, acc);
        st_f32<true>(A.q, H_, A.attn_b, mt * 16, nt * 16, acc);
      } else if (job < 256) {
        int j2 = job - 64; int nt = j2 % 48, mt = j2 / 48;
        acc = tile_mm<256>(A.h0h, H_, A.whh0h, H_, 0, mt * 16, nt * 16, acc);
        st_f32<false>(A.gh0, 768, nullptr, mt * 16, nt * 16, acc);
      } else {
        int j2 = job - 256; int nt = j2 % 48, mt = j2 / 48;
        acc = tile_mm<256>(A.h1h, H_, A.whh1h, H_, 0, mt * 16, nt * 16, acc);
        st_f32<false>(A.gh1, 768, nullptr, mt * 16, nt * 16, acc);
      }
    }
    gsync(A.bar, syncno);

    // ---- Phase B: attention scores + softmax + ctx (2 batches per WG)
    {
      int b0 = blockIdx.x * 2;
      int which = tid >> 7;
      int s = tid & 127;
      int b = b0 + which;
      const float* ep = A.enc_proj + ((size_t)(b * S_ + s)) * H_;
      const float* qq = A.q + b * H_;
      __builtin_prefetch(ep, 0, 1);
      float sacc = 0.f;
      for (int j = 0; j < H_; ++j) sacc += A.v_w[j] * tanhf(ep[j] + qq[j]);
      sc[which][s] = sacc;
      __syncthreads();
      if (tid < 2) {
        float mx = -3.0e38f;
        for (int i = 0; i < S_; ++i) mx = fmaxf(mx, sc[tid][i]);
        float sm = 0.f;
        for (int i = 0; i < S_; ++i) { float e = __expf(sc[tid][i] - mx); sc[tid][i] = e; sm += e; }
        float inv = 1.f / sm;
        for (int i = 0; i < S_; ++i) sc[tid][i] *= inv;
      }
      __syncthreads();
      for (int wb = 0; wb < 2; ++wb) {
        int bb = b0 + wb;
        float c = 0.f;
        for (int s2 = 0; s2 < S_; ++s2)
          c += sc[wb][s2] * A.enc_out[((size_t)(bb * S_ + s2)) * H_ + tid];
        _Float16 ch = (_Float16)c;
        A.a0h[bb * 288 + 1 + tid] = ch;       // dec_l0 input [y, ctx]
        A.gc [bb * 512 + 256 + tid] = ch;     // op1 input [h1, ctx]
      }
    }
    gsync(A.bar, syncno);

    // ---- Phase C: gi0 = a0h @ Wih0p^T + bih0  (K=288, 192 tiles)
    for (int job = gwave; job < 192; job += 256) {
      int nt = job % 48, mt = job / 48;
      v8f acc{};
      acc = tile_mm<288>(A.a0h, 288, A.wih0ph, 288, 0, mt * 16, nt * 16, acc);
      st_f32<true>(A.gi0, 768, A.bih0, mt * 16, nt * 16, acc);
    }
    gsync(A.bar, syncno);

    // ---- Phase D: h0 GRU update
    {
      int g = blockIdx.x * 256 + tid;
      for (int k2 = 0; k2 < 2; ++k2) {
        int idx = g + k2 * 8192;
        int b = idx >> 8, j = idx & 255;
        float r = sigm(A.gi0[b * 768 + j]       + A.gh0[b * 768 + j]       + A.bhh0[j]);
        float z = sigm(A.gi0[b * 768 + 256 + j] + A.gh0[b * 768 + 256 + j] + A.bhh0[256 + j]);
        float n = tanhf(A.gi0[b * 768 + 512 + j] +
                        r * (A.gh0[b * 768 + 512 + j] + A.bhh0[512 + j]));
        float h = (1.f - z) * n + z * A.h0[idx];
        A.h0[idx] = h;
        A.h0h[idx] = (_Float16)h;
      }
    }
    gsync(A.bar, syncno);

    // ---- Phase E: gi1 = h0@Wih1^T + bih1 (192 tiles)
    for (int job = gwave; job < 192; job += 256) {
      int nt = job % 48, mt = job / 48;
      v8f acc{};
      acc = tile_mm<256>(A.h0h, H_, A.wih1h, H_, 0, mt * 16, nt * 16, acc);
      st_f32<true>(A.gi1, 768, A.bih1, mt * 16, nt * 16, acc);
    }
    gsync(A.bar, syncno);

    // ---- Phase F: h1 GRU update (also fills op1-input h1 half of gc)
    {
      int g = blockIdx.x * 256 + tid;
      for (int k2 = 0; k2 < 2; ++k2) {
        int idx = g + k2 * 8192;
        int b = idx >> 8, j = idx & 255;
        float r = sigm(A.gi1[b * 768 + j]       + A.gh1[b * 768 + j]       + A.bhh1[j]);
        float z = sigm(A.gi1[b * 768 + 256 + j] + A.gh1[b * 768 + 256 + j] + A.bhh1[256 + j]);
        float n = tanhf(A.gi1[b * 768 + 512 + j] +
                        r * (A.gh1[b * 768 + 512 + j] + A.bhh1[512 + j]));
        float h = (1.f - z) * n + z * A.h1[idx];
        A.h1[idx] = h;
        _Float16 hh = (_Float16)h;
        A.h1h[idx] = hh;
        A.gc[b * 512 + j] = hh;
      }
    }
    gsync(A.bar, syncno);

    // ---- Phase G: y1 = gc @ op1^T + op1_b   (64 tiles, K=512)
    for (int job = gwave; job < 64; job += 256) {
      int nt = job & 15, mt = job >> 4;
      v8f acc{};
      acc = tile_mm<512>(A.gc, 512, A.op1h, 512, 0, mt * 16, nt * 16, acc);
      st_f32<true>(A.y1, H_, A.op1_b, mt * 16, nt * 16, acc);
    }
    gsync(A.bar, syncno);

    // ---- Phase H: LayerNorm(y1) -> y1h (one wave per batch row)
    if (gwave < B_) {
      int b = gwave;
      const float* yr = A.y1 + b * H_;
      float mu = 0.f;
      for (int j = 0; j < H_; ++j) mu += yr[j];
      mu *= (1.f / 256.f);
      float vr = 0.f;
      for (int j = 0; j < H_; ++j) { float d = yr[j] - mu; vr += d * d; }
      vr *= (1.f / 256.f);
      float inv = rsqrtf(vr + 1e-5f);
      int lane = tid & 31;
      for (int j = lane * 8; j < lane * 8 + 8; ++j)
        A.y1h[b * H_ + j] = (_Float16)((yr[j] - mu) * inv * A.op_g[j] + A.op_be[j]);
    }
    gsync(A.bar, syncno);

    // ---- Phase I: y2 = leaky(y1h@op2^T + op2_b)  (32 tiles, N=128)
    for (int job = gwave; job < 32; job += 256) {
      int nt = job & 7, mt = job >> 3;
      v8f acc{};
      acc = tile_mm<256>(A.y1h, H_, A.op2h, H_, 0, mt * 16, nt * 16, acc);
      st_f16_leaky(A.y2h, 128, A.op2_b, mt * 16, nt * 16, acc);
    }
    gsync(A.bar, syncno);

    // ---- Phase J: scalar head + feedback into a0h col 0
    if (blockIdx.x == 0 && tid < B_) {
      int b = tid;
      float y = A.op3_b[0];
      for (int j = 0; j < 128; ++j) y += (float)A.y2h[b * 128 + j] * A.op3_w[j];
      A.out[(size_t)b * T_ + t] = y;
      A.a0h[b * 288] = (_Float16)y;
    }
    gsync(A.bar, syncno);
  }
}

// ------------------------------- decoder init ------------------------------
__global__ void k_init_dec(const float* dh, float* h0, float* h1,
                           _Float16* h0h, _Float16* h1h,
                           _Float16* a0h, _Float16* gc) {
  int idx = blockIdx.x * 256 + threadIdx.x;
  if (idx < B_ * H_) {
    float v = dh[idx];
    int b = idx >> 8, j = idx & 255;
    h0[idx] = v; h1[idx] = v;
    h0h[idx] = (_Float16)v; h1h[idx] = (_Float16)v;
    gc[b * 512 + j] = (_Float16)v;          // h1 half of op1 input
    gc[b * 512 + 256 + j] = (_Float16)0.f;  // ctx half (overwritten per step)
  }
  if (idx < B_ * 288) a0h[idx] = (_Float16)0.f;  // y=0, ctx overwritten, pad=0
}

// ---------------------------------------------------------------------------
extern "C" void kernel_launch(void* const* d_in, const int* in_sizes, int n_in,
                              void* d_out, int out_size, void* d_ws, size_t ws_size,
                              hipStream_t stream) {
  (void)in_sizes; (void)n_in; (void)out_size; (void)ws_size;
  auto F = [&](int i) { return (const float*)d_in[i]; };

  char* base = (char*)d_ws;
  size_t off = 0;
  auto alloc = [&](size_t bytes) -> void* {
    off = (off + 255) & ~(size_t)255;
    void* p = base + off; off += bytes; return p;
  };
  auto aF = [&](size_t n) { return (float*)alloc(n * 4); };
  auto aH = [&](size_t n) { return (_Float16*)alloc(n * 2); };

  // --- workspace ---
  int* bar = (int*)alloc(sizeof(int));
  float* t1 = aF((size_t)64 * 64 * 2048);
  float* t2 = aF((size_t)64 * 128 * 1024);
  float* eda_i  = aF((size_t)64 * 64);
  float* ppg_p  = aF((size_t)64 * 64 * S_);
  float* acc_p  = aF((size_t)64 * 64 * S_);
  float* eda_p  = aF((size_t)64 * 32 * S_);
  float* prev_p = aF((size_t)64 * 128 * S_);
  _Float16* feat_h = aH((size_t)B_ * S_ * 288);
  float* ygem = aF((size_t)B_ * S_ * H_);
  _Float16* xh  = aH((size_t)B_ * S_ * H_);
  _Float16* x1h = aH((size_t)B_ * S_ * 512);
  _Float16* x2h = aH((size_t)B_ * S_ * 512);
  float* gi_f = aF((size_t)B_ * S_ * 768);
  float* gi_b = aF((size_t)B_ * S_ * 768);
  float* ghs_f = aF((size_t)B_ * 768);
  float* ghs_b = aF((size_t)B_ * 768);
  float* enc_out  = aF((size_t)B_ * S_ * H_);
  _Float16* enc_outh = aH((size_t)B_ * S_ * H_);
  float* enc_proj = aF((size_t)B_ * S_ * H_);
  _Float16* hcat_h = aH((size_t)B_ * 512);
  float* dh = aF((size_t)B_ * H_);
  // f16 weights
  _Float16* fp_h   = aH((size_t)256 * 288);
  _Float16* attn_h = aH((size_t)256 * 512);
  _Float16* ep_h   = aH((size_t)256 * 512);
  _Float16* wih_l0f_h = aH((size_t)768 * 256); _Float16* whh_l0f_h = aH((size_t)768 * 256);
  _Float16* wih_l0b_h = aH((size_t)768 * 256); _Float16* whh_l0b_h = aH((size_t)768 * 256);
  _Float16* wih_l1f_h = aH((size_t)768 * 512); _Float16* whh_l1f_h = aH((size_t)768 * 256);
  _Float16* wih_l1b_h = aH((size_t)768 * 512); _Float16* whh_l1b_h = aH((size_t)768 * 256);
  _Float16* wih_d0p_h = aH((size_t)768 * 288); _Float16* whh_d0_h  = aH((size_t)768 * 256);
  _Float16* wih_d1_h  = aH((size_t)768 * 256); _Float16* whh_d1_h  = aH((size_t)768 * 256);
  _Float16* op1_h = aH((size_t)256 * 512);
  _Float16* op2_h = aH((size_t)128 * 256);
  // decoder state
  float* qbuf = aF((size_t)B_ * H_);
  float* gh0 = aF((size_t)B_ * 768); float* gh1 = aF((size_t)B_ * 768);
  float* gi0 = aF((size_t)B_ * 768); float* gi1 = aF((size_t)B_ * 768);
  float* y1 = aF((size_t)B_ * H_);
  float* h0 = aF((size_t)B_ * H_);   float* h1 = aF((size_t)B_ * H_);
  _Float16* a0h = aH((size_t)B_ * 288);
  _Float16* gc  = aH((size_t)B_ * 512);
  _Float16* y1h = aH((size_t)B_ * H_);
  _Float16* y2h = aH((size_t)B_ * 128);
  _Float16* h0h = aH((size_t)B_ * H_);
  _Float16* h1h = aH((size_t)B_ * H_);

  hipMemsetAsync(bar, 0, sizeof(int), stream);

  auto f2h = [&](_Float16* dst, const float* src, int n) {
    k_f2h<<<(n + 255) / 256, 256, 0, stream>>>(dst, src, n);
  };
  // --- weight conversion ---
  f2h(fp_h,   F(FP_W),   256 * 288);
  f2h(attn_h, F(ATTN_W), 256 * 512);
  f2h(ep_h,   F(EP_W),   256 * 512);
  f2h(wih_l0f_h, F(L0F_WIH), 768 * 256); f2h(whh_l0f_h, F(L0F_WHH), 768 * 256);
  f2h(wih_l0b_h, F(L0B_WIH), 768 * 256); f2h(whh_l0b_h, F(L0B_WHH), 768 * 256);
  f2h(wih_l1f_h, F(L1F_WIH), 768 * 512); f2h(whh_l1f_h, F(L1F_WHH), 768 * 256);
  f2h(wih_l1b_h, F(L1B_WIH), 768 * 512); f2h(whh_l1b_h, F(L1B_WHH), 768 * 256);
  f2h(whh_d0_h, F(D0_WHH), 768 * 256);
  f2h(wih_d1_h, F(D1_WIH), 768 * 256);   f2h(whh_d1_h, F(D1_WHH), 768 * 256);
  f2h(op1_h, F(OP1_W), 256 * 512);       f2h(op2_h, F(OP2_W), 128 * 256);
  k_padw<<<(768 * 288 + 255) / 256, 256, 0, stream>>>(wih_d0p_h, F(D0_WIH), 768, 257, 288);

  // --- feature extraction (fp32 VALU, one-shot) ---
  auto conv = [&](const float* in, int pw, float* out,
                  int B, int Cin, int Cout, int L, int KW, int pad) {
    int n = B * Cout * L;
    k_conv<<<(n + 255) / 256, 256, 0, stream>>>(in, F(pw), F(pw + 1), F(pw + 2),
                                                F(pw + 3), out, B, Cin, Cout, L, KW, pad);
  };
  auto convmp = [&](const float* in, int pw, float* out,
                    int B, int Cin, int Cout, int L, int KW, int pad) {
    int n = B * Cout * (L / 2);
    k_conv_mp<<<(n + 255) / 256, 256, 0, stream>>>(in, F(pw), F(pw + 1), F(pw + 2),
                                                   F(pw + 3), out, B, Cin, Cout, L, KW, pad);
  };
  conv(F(IN_PPG), PPG_W1, t1, 64, 1, 32, 512, 7, 3);
  convmp(t1, PPG_W1 + 4, t2, 64, 32, 64, 512, 5, 2);
  k_avgpool<<<(64 * 64 * 128 + 255) / 256, 256, 0, stream>>>(t2, ppg_p, 64 * 64, 256, 2);
  conv(F(IN_ACC), ACC_W1, t1, 64, 3, 32, 256, 7, 3);
  convmp(t1, ACC_W1 + 4, acc_p, 64, 32, 64, 256, 5, 2);
  k_interp<<<(64 * 64 + 255) / 256, 256, 0, stream>>>(F(IN_EDA), eda_i, 64, 32, 64);
  conv(eda_i, EDA_W1, t1, 64, 1, 16, 64, 7, 3);
  convmp(t1, EDA_W1 + 4, t2, 64, 16, 32, 64, 5, 2);
  k_interp<<<(64 * 32 * 128 + 255) / 256, 256, 0, stream>>>(t2, eda_p, 64 * 32, 32, 128);
  conv(F(IN_PREV), PREV_W1, t1, 64, 1, 64, 2048, 7, 3);
  convmp(t1, PREV_W1 + 4, t2, 64, 64, 128, 2048, 5, 2);
  k_avgpool<<<(64 * 128 * 128 + 255) / 256, 256, 0, stream>>>(t2, prev_p, 64 * 128, 1024, 8);

  k_assemble<<<(B_ * S_ * 288 + 255) / 256, 256, 0, stream>>>(ppg_p, acc_p, eda_p, prev_p, feat_h);

  auto gemm = [&](const _Float16* A, int lda, const _Float16* W, int ldw, int wof,
                  const float* bias, float* C, _Float16* Ch, int ldc,
                  int M, int N, int K, int act) {
    int tiles = (M / 16) * (N / 16);
    int blocks = (tiles + 7) / 8;
    if (K == 256)
      k_gemm<256><<<blocks, 256, 0, stream>>>(A, lda, W, ldw, wof, bias, C, Ch, ldc, M, N, act);
    else if (K == 288)
      k_gemm<288><<<blocks, 256, 0, stream>>>(A, lda, W, ldw, wof, bias, C, Ch, ldc, M, N, act);
    else
      k_gemm<512><<<blocks, 256, 0, stream>>>(A, lda, W, ldw, wof, bias, C, Ch, ldc, M, N, act);
  };

  // --- fp projection + LN ---
  gemm(feat_h, 288, fp_h, 288, 0, F(FP_B), ygem, nullptr, H_, B_ * S_, H_, 288, 0);
  k_ln<<<(B_ * S_ + 255) / 256, 256, 0, stream>>>(ygem, F(FP_G), F(FP_BE), xh, B_ * S_, H_);

  // --- encoder layer 0 ---
  gemm(xh, H_, wih_l0f_h, H_, 0, F(L0F_BIH), gi_f, nullptr, 768, B_ * S_, 768, H_, 0);
  gemm(xh, H_, wih_l0b_h, H_, 0, F(L0B_BIH), gi_b, nullptr, 768, B_ * S_, 768, H_, 0);
  k_enc_scan<<<2, 1024, 0, stream>>>(gi_f, gi_b, whh_l0f_h, whh_l0b_h,
                                     F(L0F_BHH), F(L0B_BHH), ghs_f, ghs_b,
                                     x1h, 512, nullptr);
  // --- encoder layer 1 ---
  gemm(x1h, 512, wih_l1f_h, 512, 0, F(L1F_BIH), gi_f, nullptr, 768, B_ * S_, 768, 512, 0);
  gemm(x1h, 512, wih_l1b_h, 512, 0, F(L1B_BIH), gi_b, nullptr, 768, B_ * S_, 768, 512, 0);
  k_enc_scan<<<2, 1024, 0, stream>>>(gi_f, gi_b, whh_l1f_h, whh_l1b_h,
                                     F(L1F_BHH), F(L1B_BHH), ghs_f, ghs_b,
                                     x2h, 512, hcat_h);
  // --- encoder products ---
  gemm(x2h, 512, ep_h, 512, 0, F(EP_B), enc_out, enc_outh, H_, B_ * S_, H_, 512, 0);
  gemm(enc_outh, H_, attn_h, 512, 256, nullptr, enc_proj, nullptr, H_, B_ * S_, H_, H_, 0);
  gemm(hcat_h, 512, ep_h, 512, 0, F(EP_B), dh, nullptr, H_, B_, H_, 512, 1); // tanh
  k_init_dec<<<128, 256, 0, stream>>>(dh, h0, h1, h0h, h1h, a0h, gc);

  // --- decoder ---
  DecArgs DA;
  DA.attn_h = attn_h; DA.whh0h = whh_d0_h; DA.whh1h = whh_d1_h;
  DA.wih0ph = wih_d0p_h; DA.wih1h = wih_d1_h; DA.op1h = op1_h; DA.op2h = op2_h;
  DA.attn_b = F(ATTN_B); DA.bih0 = F(D0_BIH); DA.bhh0 = F(D0_BHH);
  DA.bih1 = F(D1_BIH); DA.bhh1 = F(D1_BHH);
  DA.op1_b = F(OP1_B); DA.op_g = F(OP_G); DA.op_be = F(OP_BE);
  DA.op2_b = F(OP2_B); DA.op3_w = F(OP3_W); DA.op3_b = F(OP3_B); DA.v_w = F(V_W);
  DA.enc_out = enc_out; DA.enc_proj = enc_proj;
  DA.q = qbuf; DA.gh0 = gh0; DA.gh1 = gh1; DA.gi0 = gi0; DA.gi1 = gi1;
  DA.y1 = y1; DA.h0 = h0; DA.h1 = h1;
  DA.a0h = a0h; DA.gc = gc;
  DA.y1h = y1h; DA.y2h = y2h; DA.h0h = h0h; DA.h1h = h1h;
  DA.out = (float*)d_out; DA.bar = bar;
  k_decoder<<<NWG, 256, 0, stream>>>(DA);
}
